// TextLSTM_38972533244245
// MI455X (gfx1250) — compile-verified, matrix operand order checked by reading, not run on hardware
//
#include <hip/hip_runtime.h>
#include <hip/hip_bf16.h>

typedef __attribute__((ext_vector_type(16))) _Float16 v16h;
typedef __attribute__((ext_vector_type(8)))  float    v8f;

#define B_   64
#define S_   200
#define E_   300
#define EP_  320
#define H_   512
#define G4_  2048
#define O_   5
#define NWG_REC 16

union Frag { uint4 u[2]; v16h h; };

__device__ inline float sigmoidf_(float x) { return 1.f / (1.f + __expf(-x)); }

// ---------------------------------------------------------------------------
// Phase 1a: embedding gather, fp32 -> f16, pad E 300 -> 320.
// Row r = s*64 + b  (time-major like the reference's transpose(emb[x], (1,0,2)))
// ---------------------------------------------------------------------------
__global__ void k_embed(const int* __restrict__ x, const float* __restrict__ emb,
                        _Float16* __restrict__ Ax) {
    int r = blockIdx.x;              // 0 .. 12799
    int s = r >> 6, b = r & 63;
    int tok = x[b * S_ + s];
    const float* src = emb + (size_t)tok * E_;
    _Float16* dst = Ax + (size_t)r * EP_;
    for (int c = threadIdx.x; c < EP_; c += blockDim.x)
        dst[c] = (c < E_) ? (_Float16)src[c] : (_Float16)0.f;
}

// ---------------------------------------------------------------------------
// Phase 1b: weight conversion (W_ih padded to K=320, W_hh as-is), bias sum.
// ---------------------------------------------------------------------------
__global__ void k_prep(const float* __restrict__ Wih, const float* __restrict__ Whh,
                       const float* __restrict__ bih, const float* __restrict__ bhh,
                       _Float16* __restrict__ Wih16, _Float16* __restrict__ Whh16,
                       float* __restrict__ bias) {
    int stride = gridDim.x * blockDim.x;
    int t0 = blockIdx.x * blockDim.x + threadIdx.x;
    for (int i = t0; i < G4_ * EP_; i += stride) {
        int n = i / EP_, k = i - n * EP_;
        Wih16[i] = (k < E_) ? (_Float16)Wih[n * E_ + k] : (_Float16)0.f;
    }
    for (int i = t0; i < G4_ * H_; i += stride) Whh16[i] = (_Float16)Whh[i];
    for (int i = t0; i < G4_;      i += stride) bias[i] = bih[i] + bhh[i];
}

// ---------------------------------------------------------------------------
// Phase 2: Xproj[s*64+b, 4H] = Ax @ Wih16^T + bias   (f16 WMMA, f32 accum)
// One 16x16 tile per wave, K = 320 (10 WMMA k-steps).
// ---------------------------------------------------------------------------
__global__ void k_gemm_in(const _Float16* __restrict__ Ax, const _Float16* __restrict__ Wih16,
                          const float* __restrict__ bias, _Float16* __restrict__ Xproj) {
    int wv = threadIdx.x >> 5, lane = threadIdx.x & 31;
    int lr = lane & 15, hi = lane >> 4;
    int T  = blockIdx.x * 8 + wv;        // global tile id
    int mt = T >> 7, nt = T & 127;       // 800 m-tiles x 128 n-tiles
    int m0 = mt * 16, n0 = nt * 16;

    v8f acc;
    float bv = bias[n0 + lr];            // D col = lane&15 for every acc slot
    #pragma unroll
    for (int v = 0; v < 8; ++v) acc[v] = bv;

    const _Float16* arow = Ax    + (size_t)(m0 + lr) * EP_ + hi * 8;
    const _Float16* brow = Wih16 + (size_t)(n0 + lr) * EP_ + hi * 8;
    #pragma unroll
    for (int kk = 0; kk < EP_; kk += 32) {
        Frag a, b;
        const uint4* pa = (const uint4*)(arow + kk);
        a.u[0] = pa[0]; a.u[1] = pa[2];            // K chunks: hi*8, 16+hi*8
        const uint4* pb = (const uint4*)(brow + kk);
        b.u[0] = pb[0]; b.u[1] = pb[2];
        acc = __builtin_amdgcn_wmma_f32_16x16x32_f16(false, a.h, false, b.h,
                                                     (short)0, acc, false, false);
    }

    _Float16* orow = Xproj + (size_t)(m0 + hi * 8) * G4_ + n0 + lr;
    #pragma unroll
    for (int v = 0; v < 8; ++v) orow[(size_t)v * G4_] = (_Float16)acc[v];
}

// ---------------------------------------------------------------------------
// Phase 3: persistent LSTM recurrence. 16 WGs x 8 waves = 128 waves.
// Wave W: m-tile = W/32 (batch rows), j-tile = W%32 (hidden cols).
// Computes i,f,g,o tiles for (m,j) -> lanes align -> in-register cell update.
// h double-buffered f16; one grid barrier per step.
// ---------------------------------------------------------------------------
__global__ void k_lstm(const _Float16* __restrict__ Xproj, const _Float16* __restrict__ Whh16,
                       float* __restrict__ c, _Float16* __restrict__ hbuf,
                       float* __restrict__ h32, unsigned int* __restrict__ cnt) {
    __shared__ __align__(16) _Float16 hA[16 * H_];   // 16 KB: h rows m0..m0+15
    int tid = threadIdx.x;
    int wv = tid >> 5, lane = tid & 31;
    int lr = lane & 15, hi = lane >> 4;
    int W  = blockIdx.x * 8 + wv;
    int m0 = (W >> 5) * 16;          // all 8 waves of a block share m0
    int j0 = (W & 31) * 16;

    for (int s = 0; s < S_; ++s) {
        const _Float16* hrd = hbuf + (size_t)(s & 1) * (B_ * H_);
        _Float16*       hwr = hbuf + (size_t)((s + 1) & 1) * (B_ * H_);

        // stage h m-tile (16 x 512 f16 = 1024 uint4) into LDS
        {
            const uint4* src = (const uint4*)(hrd + (size_t)m0 * H_);
            uint4* dst = (uint4*)hA;
            for (int i = tid; i < (16 * H_) / 8; i += blockDim.x) dst[i] = src[i];
        }
        __syncthreads();

        // C-init from precomputed input projection (+bias already folded in)
        v8f acc[4];
        const _Float16* xp = Xproj + ((size_t)s * B_ + m0) * G4_;
        #pragma unroll
        for (int g = 0; g < 4; ++g)
            #pragma unroll
            for (int v = 0; v < 8; ++v)
                acc[g][v] = (float)xp[(size_t)(v + hi * 8) * G4_ + g * H_ + j0 + lr];

        // gates += h @ W_hh^T : K = 512, 16 k-steps x 4 gate tiles
        for (int kk = 0; kk < H_; kk += 32) {
            Frag a;
            const uint4* pa = (const uint4*)(hA + lr * H_ + kk + hi * 8);
            a.u[0] = pa[0]; a.u[1] = pa[2];
            #pragma unroll
            for (int g = 0; g < 4; ++g) {
                Frag b;
                const uint4* pb = (const uint4*)(Whh16 + (size_t)(g * H_ + j0 + lr) * H_ + kk + hi * 8);
                b.u[0] = pb[0]; b.u[1] = pb[2];
                acc[g] = __builtin_amdgcn_wmma_f32_16x16x32_f16(false, a.h, false, b.h,
                                                                (short)0, acc[g], false, false);
            }
        }

        // cell update fully in registers (lanes of i,f,g,o tiles align)
        #pragma unroll
        for (int v = 0; v < 8; ++v) {
            int row = m0 + v + hi * 8;
            int col = j0 + lr;
            float iv = sigmoidf_(acc[0][v]);
            float fv = sigmoidf_(acc[1][v]);
            float gv = tanhf(acc[2][v]);
            float ov = sigmoidf_(acc[3][v]);
            size_t idx = (size_t)row * H_ + col;
            float cn = fv * c[idx] + iv * gv;
            c[idx] = cn;
            float hn = ov * tanhf(cn);
            h32[idx] = hn;
            hwr[idx] = (_Float16)hn;
        }

        // grid-wide barrier between steps
        __syncthreads();
        if (tid == 0) {
            __threadfence();
            __hip_atomic_fetch_add(cnt, 1u, __ATOMIC_RELEASE, __HIP_MEMORY_SCOPE_AGENT);
            unsigned target = (unsigned)NWG_REC * (unsigned)(s + 1);
            while (__hip_atomic_load(cnt, __ATOMIC_ACQUIRE, __HIP_MEMORY_SCOPE_AGENT) < target)
                __builtin_amdgcn_s_sleep(1);
        }
        __syncthreads();
    }
}

// ---------------------------------------------------------------------------
// Phase 4: logits = h @ W_fc^T + b_fc ; softmax. Tiny: one thread per batch row.
// ---------------------------------------------------------------------------
__global__ void k_fc(const float* __restrict__ h32, const float* __restrict__ Wfc,
                     const float* __restrict__ bfc, float* __restrict__ out) {
    int b = threadIdx.x;
    if (b >= B_) return;
    float logit[O_];
    const float* hrow = h32 + (size_t)b * H_;
    #pragma unroll
    for (int o = 0; o < O_; ++o) {
        float acc = bfc[o];
        const float* w = Wfc + (size_t)o * H_;
        for (int k = 0; k < H_; ++k) acc += hrow[k] * w[k];
        logit[o] = acc;
    }
    float mx = logit[0];
    #pragma unroll
    for (int o = 1; o < O_; ++o) mx = fmaxf(mx, logit[o]);
    float sum = 0.f;
    #pragma unroll
    for (int o = 0; o < O_; ++o) { logit[o] = __expf(logit[o] - mx); sum += logit[o]; }
    float inv = 1.f / sum;
    #pragma unroll
    for (int o = 0; o < O_; ++o) out[b * O_ + o] = logit[o] * inv;
}

// ---------------------------------------------------------------------------
extern "C" void kernel_launch(void* const* d_in, const int* in_sizes, int n_in,
                              void* d_out, int out_size, void* d_ws, size_t ws_size,
                              hipStream_t stream) {
    (void)in_sizes; (void)n_in; (void)out_size; (void)ws_size;
    const int*   x   = (const int*)  d_in[0];
    const float* emb = (const float*)d_in[1];
    const float* Wih = (const float*)d_in[2];
    const float* Whh = (const float*)d_in[3];
    const float* bih = (const float*)d_in[4];
    const float* bhh = (const float*)d_in[5];
    const float* Wfc = (const float*)d_in[6];
    const float* bfc = (const float*)d_in[7];
    float* out = (float*)d_out;

    char* ws = (char*)d_ws;
    size_t off = 0;
    auto alloc = [&](size_t bytes) {
        char* p = ws + off;
        off += (bytes + 255) & ~(size_t)255;
        return p;
    };
    _Float16* Ax    = (_Float16*)alloc((size_t)S_ * B_ * EP_ * 2);   //  8.2 MB
    _Float16* Wih16 = (_Float16*)alloc((size_t)G4_ * EP_ * 2);       //  1.3 MB
    _Float16* Whh16 = (_Float16*)alloc((size_t)G4_ * H_ * 2);        //  2.1 MB
    float*    bias  = (float*)   alloc((size_t)G4_ * 4);
    _Float16* Xproj = (_Float16*)alloc((size_t)S_ * B_ * G4_ * 2);   // 52.4 MB
    char* zbase = ws + off;                                          // zeroed region:
    float*    c     = (float*)   alloc((size_t)B_ * H_ * 4);         //   cell state
    _Float16* hbuf  = (_Float16*)alloc((size_t)2 * B_ * H_ * 2);     //   h ping-pong (f16)
    float*    h32   = (float*)   alloc((size_t)B_ * H_ * 4);         //   h (f32, for FC)
    unsigned* cnt   = (unsigned*)alloc(256);                         //   grid barrier
    size_t zbytes = (size_t)((ws + off) - zbase);

    hipMemsetAsync(zbase, 0, zbytes, stream);   // capture-safe reset each launch

    k_embed  <<<S_ * B_, 64, 0, stream>>>(x, emb, Ax);
    k_prep   <<<1024, 256, 0, stream>>>(Wih, Whh, bih, bhh, Wih16, Whh16, bias);
    k_gemm_in<<<(S_ * B_ / 16) * (G4_ / 16) / 8, 256, 0, stream>>>(Ax, Wih16, bias, Xproj);
    k_lstm   <<<NWG_REC, 256, 0, stream>>>(Xproj, Whh16, c, hbuf, h32, cnt);
    k_fc     <<<1, 64, 0, stream>>>(h32, Wfc, bfc, out);
}
